// ResidualVQ_14096082665621
// MI455X (gfx1250) — matrix operand order repaired and checked by reading, not compiled, not run on hardware
//
#include <hip/hip_runtime.h>
#include <hip/hip_bf16.h>

typedef __attribute__((ext_vector_type(16))) __bf16 v16bf;
typedef __attribute__((ext_vector_type(8)))  float  v8f;
typedef __attribute__((ext_vector_type(4)))  int    v4i;

#define BATCH 8
#define SEQ   2048
#define DIM   512
#define NUM_Q 8
#define CBSZ  1024
#define TOK   (BATCH*SEQ)      // 16384 tokens
#define WAVES 8
#define BLK   (WAVES*32)       // 256 threads
#define TPB   (WAVES*16)       // 128 tokens per block
#define NBLK  (TOK/TPB)        // 128 blocks
#define KB    (DIM/32)         // 16 K-blocks of 32
#define NT    (CBSZ/16)        // 64 code tiles of 16

// LDS padding to break bank conflicts (64 banks x 4B, 256B period):
#define RPD   (DIM + 4)        // residual row pitch in floats  (2064 B -> 4-bank lane skew)
#define PD    (DIM + 16)       // stage row pitch in bf16       (1056 B -> 8-bank skew, 32B aligned)
#define STAGE_ELEMS (16 * PD)  // one staged code tile (bf16 elems)

// ---- gfx1250 async global->LDS copy (ASYNCcnt-tracked), with safe fallback ----
#if defined(__has_builtin)
#  if __has_builtin(__builtin_amdgcn_global_load_async_to_lds_b128)
#    define HAVE_ASYNC_LDS 1
#  endif
#endif
#ifndef HAVE_ASYNC_LDS
#  define HAVE_ASYNC_LDS 0
#endif

typedef __attribute__((address_space(1))) v4i gv4i;   // global int4
typedef __attribute__((address_space(3))) v4i lv4i;   // LDS int4

__device__ __forceinline__ void async_cp16(const void* g, void* l) {
#if HAVE_ASYNC_LDS
  __builtin_amdgcn_global_load_async_to_lds_b128((gv4i*)g, (lv4i*)l, 0, 0);
#else
  *(float4*)l = *(const float4*)g;   // sync fallback (vmem load + ds store)
#endif
}

__device__ __forceinline__ void wait_async_all() {
  asm volatile("s_wait_asynccnt 0x0" ::: "memory");
}

// Stage one 16-code x 512-dim bf16 tile (16 KB) into padded LDS rows.
// 256 threads: thread = (row 0..15) x (64B segment 0..15); 4 x b128 each.
__device__ __forceinline__ void stage_tile(const __bf16* __restrict__ gsrc,
                                           __bf16* ldst, int tid) {
  const int r = tid >> 4;
  const int s = tid & 15;
  const char* g = (const char*)gsrc + (size_t)r * (DIM * 2) + s * 64;
  char*       l = (char*)ldst       + (size_t)r * (PD  * 2) + s * 64;
  #pragma unroll
  for (int j = 0; j < 4; ++j) async_cp16(g + 16 * j, l + 16 * j);
}

// ---------------------------------------------------------------------------
// Prep: codebooks f32 -> bf16 (WMMA B operand) + per-code squared norms.
// ---------------------------------------------------------------------------
__global__ void rvq_prep(const float* __restrict__ cbf,
                         __bf16* __restrict__ cbb,
                         float* __restrict__ norms) {
  const int wave = threadIdx.x >> 5;
  const int lane = threadIdx.x & 31;
  const int code = blockIdx.x * WAVES + wave;          // 0 .. NUM_Q*CBSZ-1
  const float* src = cbf + (size_t)code * DIM;
  __bf16*      dst = cbb + (size_t)code * DIM;
  float nrm = 0.f;
  #pragma unroll
  for (int j = 0; j < 16; ++j) {
    int i = lane * 16 + j;
    float v = src[i];
    nrm += v * v;
    dst[i] = (__bf16)v;
  }
  #pragma unroll
  for (int m = 16; m >= 1; m >>= 1) nrm += __shfl_xor(nrm, m, 32);
  if (lane == 0) norms[code] = nrm;
}

// ---------------------------------------------------------------------------
// Main: per-wave 16-token tile; f32 residual LDS-resident across all layers.
// B tiles staged global->LDS asynchronously, double-buffered, shared by 8 waves.
// ---------------------------------------------------------------------------
__global__ void __launch_bounds__(BLK, 1) rvq_main(
    const float* __restrict__ x,      const float* __restrict__ cbf,
    const __bf16* __restrict__ cbb,   const float* __restrict__ norms,
    float* __restrict__ qout,         float* __restrict__ idxout,
    float* __restrict__ lossws) {
  extern __shared__ char lds_raw[];
  float*  resid  = (float*)lds_raw;                               // [128][RPD] f32
  __bf16* bstage = (__bf16*)(lds_raw + (size_t)TPB * RPD * 4);    // 2 x [16][PD] bf16
  int*    idxs   = (int*)((char*)bstage + 2 * STAGE_ELEMS * 2);   // [8][16]
  float*  lossp  = (float*)((char*)idxs + WAVES * 16 * 4);        // [8][8]

  const int tid  = threadIdx.x;
  const int wave = tid >> 5;
  const int lane = tid & 31;
  const int hh   = lane >> 4;              // half-wave (0/1)
  const int l16  = lane & 15;
  const int tok0 = blockIdx.x * TPB;
  const int wtok = tok0 + wave * 16;

  // Stage x -> LDS residual (padded rows).
  for (int i = tid; i < TPB * (DIM / 4); i += BLK) {
    int row = i / (DIM / 4), c4 = i % (DIM / 4);
    *(float4*)(resid + (size_t)row * RPD + c4 * 4) =
        *(const float4*)(x + ((size_t)(tok0 + row)) * DIM + c4 * 4);
  }
  __syncthreads();

  float* wres = resid + (size_t)wave * 16 * RPD;

  for (int q = 0; q < NUM_Q; ++q) {
    // Build all 16 A-fragments (16x32 bf16, contiguous k = 16*hh + s order) from f32 residual.
    v16bf a[KB];
    #pragma unroll
    for (int kb = 0; kb < KB; ++kb) {
      const float* rp = wres + (size_t)l16 * RPD + kb * 32 + hh * 16;
      #pragma unroll
      for (int j = 0; j < 8; ++j) {
        a[kb][j]     = (__bf16)rp[j];
        a[kb][8 + j] = (__bf16)rp[8 + j];
      }
    }

    const __bf16* cq = cbb + (size_t)q * CBSZ * DIM;
    const float*  nq = norms + q * CBSZ;

    float bestv[8];
    int   besti[8];
    #pragma unroll
    for (int r = 0; r < 8; ++r) { bestv[r] = 3.4e38f; besti[r] = 0; }

    // Prime the pipeline: stage tile 0.
    stage_tile(cq, bstage, tid);
    wait_async_all();
    __syncthreads();

    #pragma unroll 1
    for (int ct = 0; ct < NT; ++ct) {
      const __bf16* cur = bstage + (ct & 1) * STAGE_ELEMS;
      if (ct + 1 < NT)            // overlap next tile's DMA with this tile's WMMAs
        stage_tile(cq + (size_t)(ct + 1) * 16 * DIM,
                   bstage + ((ct + 1) & 1) * STAGE_ELEMS, tid);

      const int   code = ct * 16 + l16;                 // this lane's column N
      const float nrm  = nq[code];
      const __bf16* bp = cur + (size_t)l16 * PD + hh * 16;
      v8f acc = {};
      #pragma unroll
      for (int kb = 0; kb < KB; ++kb) {
        v16bf b = *(const v16bf*)(bp + kb * 32);        // 32B LDS read, B layout
        acc = __builtin_amdgcn_wmma_f32_16x16x32_bf16(
            false, a[kb], false, b, (short)0, acc, false, false);
      }
      #pragma unroll
      for (int r = 0; r < 8; ++r) {                     // row M = r + 8*hh
        float s = nrm - 2.0f * acc[r];
        if (s < bestv[r]) { bestv[r] = s; besti[r] = code; }
      }
      wait_async_all();          // next buffer landed
      __syncthreads();           // everyone done reading cur; next tile visible
    }

    // Argmin across the 16 lanes of each half (N dimension); ties -> low idx.
    #pragma unroll
    for (int r = 0; r < 8; ++r) {
      #pragma unroll
      for (int m = 8; m >= 1; m >>= 1) {
        float ov = __shfl_xor(bestv[r], m, 32);
        int   oi = __shfl_xor(besti[r], m, 32);
        if (ov < bestv[r] || (ov == bestv[r] && oi < besti[r])) {
          bestv[r] = ov; besti[r] = oi;
        }
      }
      if (l16 == 0) idxs[wave * 16 + hh * 8 + r] = besti[r];
    }
    __syncthreads();

    // Residual update + commit loss: mean((quant - resid)^2) = mean(new_resid^2).
    float lacc = 0.f;
    #pragma unroll 1
    for (int m = 0; m < 16; ++m) {
      const int code = idxs[wave * 16 + m];
      const float4* cp4 = (const float4*)(cbf + ((size_t)q * CBSZ + code) * DIM);
      float* rp = wres + (size_t)m * RPD;
      #pragma unroll
      for (int i = lane; i < DIM / 4; i += 32) {        // 4 iters, conflict-free
        float4 c = cp4[i];
        float4 r = *(float4*)(rp + i * 4);
        r.x -= c.x; r.y -= c.y; r.z -= c.z; r.w -= c.w;
        lacc += r.x * r.x + r.y * r.y + r.z * r.z + r.w * r.w;
        *(float4*)(rp + i * 4) = r;
      }
    }
    if (lane < 16)
      idxout[(size_t)(wtok + lane) * NUM_Q + q] = (float)idxs[wave * 16 + lane];

    #pragma unroll
    for (int m = 16; m >= 1; m >>= 1) lacc += __shfl_xor(lacc, m, 32);
    if (lane == 0) lossp[wave * NUM_Q + q] = lacc;
  }

  __syncthreads();
  // quantized_out = x - residual_final (STE sum of selected codes).
  for (int i = tid; i < TPB * (DIM / 4); i += BLK) {
    int row = i / (DIM / 4), c4 = i % (DIM / 4);
    float4 xv = *(const float4*)(x + ((size_t)(tok0 + row)) * DIM + c4 * 4);
    float4 rv = *(const float4*)(resid + (size_t)row * RPD + c4 * 4);
    float4 o;
    o.x = xv.x - rv.x; o.y = xv.y - rv.y; o.z = xv.z - rv.z; o.w = xv.w - rv.w;
    *(float4*)(qout + ((size_t)(tok0 + row)) * DIM + c4 * 4) = o;
  }
  // Fixed-order per-block loss partials (deterministic; no float atomics).
  if (tid == 0) {
    #pragma unroll
    for (int q = 0; q < NUM_Q; ++q) {
      float s = 0.f;
      for (int w = 0; w < WAVES; ++w) s += lossp[w * NUM_Q + q];
      lossws[blockIdx.x * NUM_Q + q] = s;
    }
  }
}

// Fixed-order final loss reduction over 128 block partials.
__global__ void rvq_loss(const float* __restrict__ lossws,
                         float* __restrict__ lossout) {
  int q = threadIdx.x;
  if (q < NUM_Q) {
    float s = 0.f;
    for (int b = 0; b < NBLK; ++b) s += lossws[b * NUM_Q + q];
    lossout[q] = s * (1.0f / ((float)TOK * (float)DIM));
  }
}

extern "C" void kernel_launch(void* const* d_in, const int* in_sizes, int n_in,
                              void* d_out, int out_size, void* d_ws, size_t ws_size,
                              hipStream_t stream) {
  const float* x   = (const float*)d_in[0];   // [8,2048,512] f32
  const float* cbf = (const float*)d_in[1];   // [8,1024,512] f32

  float* qout    = (float*)d_out;                          // [8,2048,512]
  float* idxout  = qout + (size_t)TOK * DIM;               // [8,2048,8] (as f32)
  float* lossout = idxout + (size_t)TOK * NUM_Q;           // [8]

  // Workspace: bf16 codebooks (8 MB) | code norms (32 KB) | block loss partials (4 KB)
  __bf16* cbb    = (__bf16*)d_ws;
  float*  norms  = (float*)((char*)d_ws + (size_t)NUM_Q * CBSZ * DIM * 2);
  float*  lossws = norms + NUM_Q * CBSZ;

  rvq_prep<<<NUM_Q * CBSZ / WAVES, BLK, 0, stream>>>(cbf, cbb, norms);

  size_t shmem = (size_t)TPB * RPD * 4            // residual (padded)
               + 2 * (size_t)STAGE_ELEMS * 2      // double-buffered B stage
               + WAVES * 16 * 4 + WAVES * NUM_Q * 4;   // ~292 KB < 320 KB WGP
  rvq_main<<<NBLK, BLK, shmem, stream>>>(x, cbf, cbb, norms, qout, idxout, lossws);

  rvq_loss<<<1, 32, 0, stream>>>(lossws, lossout);
}
